// ScratchLSTM_43911745634374
// MI455X (gfx1250) — matrix operand order repaired and checked by reading, not compiled, not run on hardware
//
#include <hip/hip_runtime.h>
#include <hip/hip_bf16.h>
#include <stdint.h>

// Problem dims (fixed by the reference)
#define T_DIM 256
#define B_DIM 128
#define F_DIM 1024
#define H_DIM 1024
#define G4H   4096   // 4*H

typedef __attribute__((ext_vector_type(16))) __bf16 v16bf;
typedef __attribute__((ext_vector_type(8)))  float  v8f;
typedef __attribute__((ext_vector_type(4)))  float  f32x4;

union BfVec {
  v16bf v;
  f32x4 f[2];
};

__device__ __forceinline__ unsigned short f2bf(float x) {
  unsigned int u = __float_as_uint(x);
  u += 0x7FFFu + ((u >> 16) & 1u);        // round-to-nearest-even
  return (unsigned short)(u >> 16);
}

__device__ __forceinline__ float fast_sigmoid(float x) {
  return 1.f / (1.f + __expf(-x));
}
__device__ __forceinline__ float fast_tanh(float x) {
  // 1 - 2/(e^{2x}+1); saturates correctly at +/-1 for large |x|
  return 1.f - 2.f / (__expf(2.f * x) + 1.f);
}

// ---------------------------------------------------------------------------
// Core WMMA tile: one wave accumulates 4 x (16x16) fp32 tiles.
// A  : M x K row-major bf16 (as ushort)
// Br0..3 : per-wave row pointers into the N x K (transposed) weight matrix.
// Fragment addressing (contiguous k order used here, same for A and B):
//   A (16-bit 16x32): lane m=lane&15; halves 0..15 -> K=kb+0..15,
//                     kb = (lane>=16) ? 16 : 0  => two 16B loads.
//   B (16-bit 32x16): lane n=lane&15; halves 0..15 -> K=kb+0..15,
//                     kb = (lane>=16) ? 16 : 0  => two 16B loads (32B contiguous).
// ---------------------------------------------------------------------------
__device__ __forceinline__ void wmma_k_loop(const unsigned short* __restrict__ Arow,
                                            const unsigned short* __restrict__ Br0,
                                            const unsigned short* __restrict__ Br1,
                                            const unsigned short* __restrict__ Br2,
                                            const unsigned short* __restrict__ Br3,
                                            int K, int aoff, int boff,
                                            v8f acc[4]) {
  #pragma unroll 4
  for (int k = 0; k < K; k += 32) {
    BfVec a;
    a.f[0] = *(const f32x4*)(Arow + k + aoff);
    a.f[1] = *(const f32x4*)(Arow + k + aoff + 8);

    BfVec b0, b1, b2, b3;
    b0.f[0] = *(const f32x4*)(Br0 + k + boff);
    b0.f[1] = *(const f32x4*)(Br0 + k + boff + 8);
    b1.f[0] = *(const f32x4*)(Br1 + k + boff);
    b1.f[1] = *(const f32x4*)(Br1 + k + boff + 8);
    b2.f[0] = *(const f32x4*)(Br2 + k + boff);
    b2.f[1] = *(const f32x4*)(Br2 + k + boff + 8);
    b3.f[0] = *(const f32x4*)(Br3 + k + boff);
    b3.f[1] = *(const f32x4*)(Br3 + k + boff + 8);

    acc[0] = __builtin_amdgcn_wmma_f32_16x16x32_bf16(false, a.v, false, b0.v,
                                                     (short)0, acc[0], false, false);
    acc[1] = __builtin_amdgcn_wmma_f32_16x16x32_bf16(false, a.v, false, b1.v,
                                                     (short)0, acc[1], false, false);
    acc[2] = __builtin_amdgcn_wmma_f32_16x16x32_bf16(false, a.v, false, b2.v,
                                                     (short)0, acc[2], false, false);
    acc[3] = __builtin_amdgcn_wmma_f32_16x16x32_bf16(false, a.v, false, b3.v,
                                                     (short)0, acc[3], false, false);
  }
}

// ---------------------------------------------------------------------------
// Precompute GEMM: Gpre[m][n] = sum_k Xbf[m][k] * WtIH[n][k] + bias[n]
// Grid: (M/16, N/256), block 128 (4 waves, each wave owns a 16x64 N-slice).
// ---------------------------------------------------------------------------
__global__ void gemm_x_wih_kernel(const unsigned short* __restrict__ Xbf,
                                  const unsigned short* __restrict__ WtIH,
                                  const float* __restrict__ bias,
                                  float* __restrict__ Gpre) {
  v8f acc[4] = {};
  const int Mbase = blockIdx.x * 16;
  const int wave  = threadIdx.x >> 5;
  const int Nwave = blockIdx.y * 256 + wave * 64;

  const int lane = threadIdx.x & 31;
  const int mlo  = lane & 15;
  const int hi   = (lane & 16) ? 1 : 0;
  const int aoff = hi ? 16 : 0;
  const int boff = hi ? 16 : 0;

  const unsigned short* Arow = Xbf + (size_t)(Mbase + mlo) * F_DIM;
  wmma_k_loop(Arow,
              WtIH + (size_t)(Nwave +  0 + mlo) * F_DIM,
              WtIH + (size_t)(Nwave + 16 + mlo) * F_DIM,
              WtIH + (size_t)(Nwave + 32 + mlo) * F_DIM,
              WtIH + (size_t)(Nwave + 48 + mlo) * F_DIM,
              F_DIM, aoff, boff, acc);

  const int col   = lane & 15;
  const int rbase = hi ? 8 : 0;
  #pragma unroll
  for (int j = 0; j < 4; ++j) {
    #pragma unroll
    for (int i = 0; i < 8; ++i) {
      const int m = Mbase + rbase + i;
      const int n = Nwave + j * 16 + col;
      Gpre[(size_t)m * G4H + n] = acc[j][i] + bias[n];
    }
  }
}

// ---------------------------------------------------------------------------
// Fused recurrent step:
//   gates = Gpre_t + h_{t-1} @ W_hh   (WMMA, one 16-col tile x 4 gates per wave)
//   c_t = sig(f)*c + sig(i)*tanh(g);  h_t = sig(o)*tanh(c_t)
// Writes: hseq_t (fp32), HbfNext (bf16, next step's A), Cst; optionally h/c finals.
// Grid: (B/16, H/64), block 128 (4 waves, wave w owns cols (by*4+w)*16 .. +15).
// ---------------------------------------------------------------------------
__global__ void lstm_step_fused_kernel(const unsigned short* __restrict__ Hbf,
                                       const unsigned short* __restrict__ WtHH,
                                       const float* __restrict__ Gpre_t,
                                       float* __restrict__ Cst,
                                       unsigned short* __restrict__ HbfNext,
                                       float* __restrict__ hseq_t,
                                       float* __restrict__ hfin,
                                       float* __restrict__ cfin) {
  v8f acc[4] = {};
  const int Mbase = blockIdx.x * 16;
  const int wave  = threadIdx.x >> 5;
  const int n0    = (blockIdx.y * 4 + wave) * 16;   // column tile within H

  const int lane = threadIdx.x & 31;
  const int mlo  = lane & 15;
  const int hi   = (lane & 16) ? 1 : 0;
  const int aoff = hi ? 16 : 0;
  const int boff = hi ? 16 : 0;
  const int col  = lane & 15;
  const int rbase = hi ? 8 : 0;

  // Prefetch the precomputed-gate operands needed in the epilogue so the
  // 512MB Gpre stream overlaps the WMMA loop (lowers to global_prefetch_b8).
  {
    const size_t go = (size_t)(Mbase + rbase) * G4H + n0 + col;
    __builtin_prefetch(Gpre_t + go,             0, 1);
    __builtin_prefetch(Gpre_t + go + H_DIM,     0, 1);
    __builtin_prefetch(Gpre_t + go + 2 * H_DIM, 0, 1);
    __builtin_prefetch(Gpre_t + go + 3 * H_DIM, 0, 1);
  }

  const unsigned short* Arow = Hbf + (size_t)(Mbase + mlo) * H_DIM;
  // Same 16 columns for all four gates: rows n0+g*H of the transposed W_hh.
  wmma_k_loop(Arow,
              WtHH + (size_t)(n0 + 0 * H_DIM + mlo) * H_DIM,
              WtHH + (size_t)(n0 + 1 * H_DIM + mlo) * H_DIM,
              WtHH + (size_t)(n0 + 2 * H_DIM + mlo) * H_DIM,
              WtHH + (size_t)(n0 + 3 * H_DIM + mlo) * H_DIM,
              H_DIM, aoff, boff, acc);

  #pragma unroll
  for (int i = 0; i < 8; ++i) {
    const int m = Mbase + rbase + i;
    const int n = n0 + col;
    const size_t go = (size_t)m * G4H + n;

    const float zi = acc[0][i] + Gpre_t[go];
    const float zf = acc[1][i] + Gpre_t[go + H_DIM];
    const float zg = acc[2][i] + Gpre_t[go + 2 * H_DIM];
    const float zo = acc[3][i] + Gpre_t[go + 3 * H_DIM];

    const float ig = fast_sigmoid(zi);
    const float fg = fast_sigmoid(zf);
    const float gg = fast_tanh(zg);
    const float og = fast_sigmoid(zo);

    const size_t ho = (size_t)m * H_DIM + n;
    const float cn = fg * Cst[ho] + ig * gg;
    const float h  = og * fast_tanh(cn);

    Cst[ho]      = cn;
    HbfNext[ho]  = f2bf(h);
    hseq_t[ho]   = h;
    if (hfin != nullptr) { hfin[ho] = h; cfin[ho] = cn; }
  }
}

// ---------------------------------------------------------------------------
// One-time prep kernels
// ---------------------------------------------------------------------------
__global__ void cvt_f32_to_bf16_kernel(const float* __restrict__ src,
                                       unsigned short* __restrict__ dst) {
  const size_t idx = (size_t)blockIdx.x * blockDim.x + threadIdx.x;
  dst[idx] = f2bf(src[idx]);
}

// W: (K=1024) x (N=4096) fp32 row-major  ->  Wt: N x K bf16 row-major
__global__ void transpose_w_kernel(const float* __restrict__ W,
                                   unsigned short* __restrict__ Wt) {
  const size_t idx = (size_t)blockIdx.x * blockDim.x + threadIdx.x; // over N*K
  const int k = (int)(idx & 1023);   // K = 1024
  const int n = (int)(idx >> 10);
  Wt[idx] = f2bf(W[(size_t)k * G4H + n]);
}

__global__ void init_hc_kernel(const float* __restrict__ h0,
                               const float* __restrict__ c0,
                               unsigned short* __restrict__ hbf,
                               float* __restrict__ c) {
  const int idx = blockIdx.x * blockDim.x + threadIdx.x;  // B*H
  const int n = idx & (H_DIM - 1);
  hbf[idx] = f2bf(h0[n]);
  c[idx]   = c0[n];
}

// ---------------------------------------------------------------------------
// Workspace layout (bytes)
// ---------------------------------------------------------------------------
static const size_t OFF_GPRE  = 0;                                        // T*B*4H fp32
static const size_t OFF_WTIH  = OFF_GPRE  + (size_t)T_DIM * B_DIM * G4H * 4;
static const size_t OFF_WTHH  = OFF_WTIH  + (size_t)G4H * F_DIM * 2;
static const size_t OFF_XBF   = OFF_WTHH  + (size_t)G4H * H_DIM * 2;
static const size_t OFF_HBF0  = OFF_XBF   + (size_t)T_DIM * B_DIM * F_DIM * 2;
static const size_t OFF_HBF1  = OFF_HBF0  + (size_t)B_DIM * H_DIM * 2;
static const size_t OFF_C     = OFF_HBF1  + (size_t)B_DIM * H_DIM * 2;

extern "C" void kernel_launch(void* const* d_in, const int* in_sizes, int n_in,
                              void* d_out, int out_size, void* d_ws, size_t ws_size,
                              hipStream_t stream) {
  const float* x    = (const float*)d_in[0];   // (T,B,F)
  const float* h0   = (const float*)d_in[1];   // (H,)
  const float* c0   = (const float*)d_in[2];   // (H,)
  const float* Wih  = (const float*)d_in[3];   // (F,4H)
  const float* Whh  = (const float*)d_in[4];   // (H,4H)
  const float* bias = (const float*)d_in[5];   // (4H,)

  float* out = (float*)d_out;                  // hidden_seq | h_t | c_t

  char* ws = (char*)d_ws;
  float*          Gpre = (float*)(ws + OFF_GPRE);
  unsigned short* WtIH = (unsigned short*)(ws + OFF_WTIH);
  unsigned short* WtHH = (unsigned short*)(ws + OFF_WTHH);
  unsigned short* Xbf  = (unsigned short*)(ws + OFF_XBF);
  unsigned short* Hbf0 = (unsigned short*)(ws + OFF_HBF0);
  unsigned short* Hbf1 = (unsigned short*)(ws + OFF_HBF1);
  float*          Cst  = (float*)(ws + OFF_C);

  // --- one-time prep -------------------------------------------------------
  {
    const size_t nx = (size_t)T_DIM * B_DIM * F_DIM;
    cvt_f32_to_bf16_kernel<<<(unsigned)(nx / 256), 256, 0, stream>>>(x, Xbf);

    const size_t nw = (size_t)G4H * F_DIM;   // same count for both weights
    transpose_w_kernel<<<(unsigned)(nw / 256), 256, 0, stream>>>(Wih, WtIH);
    transpose_w_kernel<<<(unsigned)(nw / 256), 256, 0, stream>>>(Whh, WtHH);

    init_hc_kernel<<<(B_DIM * H_DIM) / 256, 256, 0, stream>>>(h0, c0, Hbf0, Cst);
  }

  // --- precompute Gpre = X @ W_ih + bias  (fully parallel over T) ----------
  {
    dim3 grid((T_DIM * B_DIM) / 16, G4H / 256);
    gemm_x_wih_kernel<<<grid, 128, 0, stream>>>(Xbf, WtIH, bias, Gpre);
  }

  // --- recurrence: one fused kernel per step, ping-pong bf16 h state -------
  for (int t = 0; t < T_DIM; ++t) {
    unsigned short* hsrc = (t & 1) ? Hbf1 : Hbf0;
    unsigned short* hdst = (t & 1) ? Hbf0 : Hbf1;

    const bool last = (t == T_DIM - 1);
    float* hfin = last ? out + (size_t)T_DIM * B_DIM * H_DIM : nullptr;
    float* cfin = last ? out + (size_t)T_DIM * B_DIM * H_DIM + (size_t)B_DIM * H_DIM
                       : nullptr;

    dim3 grid(B_DIM / 16, H_DIM / 64);
    lstm_step_fused_kernel<<<grid, 128, 0, stream>>>(
        hsrc, WtHH, Gpre + (size_t)t * B_DIM * G4H, Cst,
        hdst, out + (size_t)t * B_DIM * H_DIM, hfin, cfin);
  }
}
